// Attention_44590350467239
// MI455X (gfx1250) — matrix operand order repaired and checked by reading, not compiled, not run on hardware
//
#include <hip/hip_runtime.h>
#include <stdint.h>

// Problem constants (fixed by the reference)
#define Nb 16
#define Lq 2048
#define Sk 2048
#define Dh 128
#define BM 128           // query rows per block (8 waves x 16)
#define BN 64            // key columns per S-tile
#define SCALE 0.08838834764831845f   // 1/sqrt(128)

typedef _Float16 h16;
typedef __attribute__((ext_vector_type(16))) _Float16     v16h;
typedef __attribute__((ext_vector_type(8)))  float        v8f;
typedef __attribute__((ext_vector_type(4)))  unsigned int v4u;
typedef __attribute__((ext_vector_type(8)))  int          v8i;
typedef __attribute__((ext_vector_type(4)))  int          v4i;

// LDS layout (bytes). Pitches padded so 16 rows map to distinct bank groups
// for ds_load_b128 (272B = 68 dwords -> 4-bank shift/row; 144B = 36 dwords).
#define KT_PITCH 272
#define VT_PITCH 144
#define PT_PITCH 144
#define KT_OFF   0
#define VT_OFF   17408                  // 64 * 272
#define PT_OFF   (VT_OFF + 18432)       // 128 * 144
#define MASK_OFF (PT_OFF + 18432)       // 8 waves * 16 * 144
#define SMEM_BYTES (MASK_OFF + BM * BN) // + 128*64 mask bytes = 62464

union H4  { h16 h[4]; uint64_t u; };
union V16 { uint4 q[2]; v16h v; };

__device__ __forceinline__ v16h cvt16(float4 a0, float4 a1, float4 b0, float4 b1) {
  v16h x;
  x[0]=(h16)a0.x;  x[1]=(h16)a0.y;  x[2]=(h16)a0.z;  x[3]=(h16)a0.w;
  x[4]=(h16)a1.x;  x[5]=(h16)a1.y;  x[6]=(h16)a1.z;  x[7]=(h16)a1.w;
  x[8]=(h16)b0.x;  x[9]=(h16)b0.y;  x[10]=(h16)b0.z; x[11]=(h16)b0.w;
  x[12]=(h16)b1.x; x[13]=(h16)b1.y; x[14]=(h16)b1.z; x[15]=(h16)b1.w;
  return x;
}

// Stage 64x128 f32 K tile -> f16 row-major LDS (pitch 272B). 256 threads.
__device__ __forceinline__ void stage_k(uint8_t* smem, const float* Kt, int t) {
  const int kr = t >> 2, q4 = t & 3;
  const float4* src = (const float4*)(Kt + (size_t)kr * Dh + q4 * 32);
  uint8_t* dst = smem + KT_OFF + kr * KT_PITCH + q4 * 64;
#pragma unroll
  for (int i = 0; i < 8; ++i) {
    float4 v = src[i];
    H4 p; p.h[0]=(h16)v.x; p.h[1]=(h16)v.y; p.h[2]=(h16)v.z; p.h[3]=(h16)v.w;
    *(uint64_t*)(dst + i * 8) = p.u;
  }
}

// Stage 64x128 f32 V tile -> f16 TRANSPOSED (d-major: 128 rows x 64 s cols,
// pitch 144B). Lane-coalesced global reads (consecutive d across lanes).
__device__ __forceinline__ void stage_v(uint8_t* smem, const float* Vt, int t) {
  const int d = t & 127, sh = t >> 7;
  const float* src = Vt + (size_t)(sh * 32) * Dh + d;
  uint8_t* dst = smem + VT_OFF + d * VT_PITCH + sh * 64;
#pragma unroll
  for (int i = 0; i < 8; ++i) {
    H4 p;
    p.h[0]=(h16)src[(size_t)(i*4+0)*Dh];
    p.h[1]=(h16)src[(size_t)(i*4+1)*Dh];
    p.h[2]=(h16)src[(size_t)(i*4+2)*Dh];
    p.h[3]=(h16)src[(size_t)(i*4+3)*Dh];
    *(uint64_t*)(dst + i * 8) = p.u;
  }
}

// TDM: DMA a 64(s) x 128(l) u8 mask tile (row stride Sk) into LDS.
// D# built per CDNA5 ISA: group0 {count, lds_addr,
// global_addr, type=2}; group1 {data_size=1B, tensor dims, tile dims, stride0}.
__device__ __forceinline__ void tdm_mask(const uint8_t* g, uint32_t lds_off) {
  const uint64_t ga = (uint64_t)(uintptr_t)g;
  v4u g0;
  g0[0] = 1u;                                   // count=1, user mode
  g0[1] = lds_off;                              // LDS byte address
  g0[2] = (unsigned)ga;                         // global_addr[31:0]
  g0[3] = (unsigned)((ga >> 32) & 0x1FFFFFFu) | (2u << 30);  // addr[56:32], type=2
  v8i g1;
  g1[0] = 0;                                    // wg_mask=0, data_size=0 (1B)
  g1[1] = (int)((Sk & 0xFFFF) << 16);           // tensor_dim0[15:0] @ bits 63:48
  g1[2] = (int)(((unsigned)Sk >> 16) | ((Lq & 0xFFFF) << 16)); // dim0 hi | dim1 lo
  g1[3] = (int)(((unsigned)Lq >> 16) | (BN << 16));            // dim1 hi | tile_dim0=64
  g1[4] = BM;                                   // tile_dim1=128, tile_dim2=0
  g1[5] = Sk;                                   // tensor_dim0_stride[31:0]
  g1[6] = 0;
  g1[7] = 0;
  v4i z4 = {0, 0, 0, 0};
#if __clang_major__ >= 23
  v8i z8 = {0, 0, 0, 0, 0, 0, 0, 0};
  __builtin_amdgcn_tensor_load_to_lds(g0, g1, z4, z4, z8, 0);
#else
  __builtin_amdgcn_tensor_load_to_lds(g0, g1, z4, z4, 0);
#endif
}

__global__ __launch_bounds__(256) void
attn_fwd_kernel(const float* __restrict__ Q, const float* __restrict__ K,
                const float* __restrict__ V, const uint8_t* __restrict__ M,
                float* __restrict__ Xo, float* __restrict__ Wo) {
  __shared__ __align__(16) uint8_t smem[SMEM_BYTES];
  const int t    = threadIdx.x;
  const int w    = t >> 5;        // wave id 0..7
  const int lane = t & 31;
  const int h    = lane >> 4;     // half-wave
  const int ln   = lane & 15;     // row (A) / col (B,D) within fragment
  const int bn   = blockIdx.x >> 4;
  const int q0   = (blockIdx.x & 15) * BM;
  const int qrow = q0 + w * 16;

  const float*   Kb = K + (size_t)bn * Sk * Dh;
  const float*   Vb = V + (size_t)bn * Sk * Dh;
  const uint8_t* Mb = M + ((size_t)bn * Lq + q0) * Sk;
  const uint32_t mask_lds = (uint32_t)(uintptr_t)(smem + MASK_OFF);

  // ---- Q tile -> 4 f16 A-fragments (K-chunks of 32 over D=128), ISA A-layout:
  // lane(h,r) elem e<8 -> dim d0+8h+e ; e>=8 -> dim d0+16+8h+(e-8)
  v16h qa[4];
  {
    const float* qr = Q + ((size_t)bn * Lq + qrow + ln) * Dh;
#pragma unroll
    for (int f = 0; f < 4; ++f) {
      const int d0 = f * 32 + 8 * h;
      qa[f] = cvt16(*(const float4*)(qr + d0),      *(const float4*)(qr + d0 + 4),
                    *(const float4*)(qr + d0 + 16), *(const float4*)(qr + d0 + 20));
    }
  }

  // ================= Pass 1: row sums of exp(scaled scores) ================
  float lsum[8];
#pragma unroll
  for (int j = 0; j < 8; ++j) lsum[j] = 0.f;

  for (int it = 0; it < Sk / BN; ++it) {
    const int s0 = it * BN;
    __syncthreads();
    if (t < 32) tdm_mask(Mb + s0, mask_lds);             // async DMA, wave 0
    stage_k(smem, Kb + (size_t)s0 * Dh, t);
    if (t < 32) __builtin_amdgcn_s_wait_tensorcnt(0);
    __syncthreads();

#pragma unroll
    for (int nt = 0; nt < 4; ++nt) {
      v8f c = {};
      const uint8_t* krow = smem + KT_OFF + (nt * 16 + ln) * KT_PITCH + 16 * h;
#pragma unroll
      for (int kc = 0; kc < 4; ++kc) {
        V16 ub;
        ub.q[0] = *(const uint4*)(krow + kc * 64);
        ub.q[1] = *(const uint4*)(krow + kc * 64 + 32);
        c = __builtin_amdgcn_wmma_f32_16x16x32_f16(false, qa[kc], false, ub.v,
                                                   (short)0, c, false, false);
      }
      const uint8_t* mp = smem + MASK_OFF + (w * 16 + 8 * h) * BN + nt * 16 + ln;
#pragma unroll
      for (int j = 0; j < 8; ++j)
        lsum[j] += mp[j * BN] ? 0.f : __expf(c[j] * SCALE);
    }
  }

  // Reduce row sums across the 16 lanes of each half-wave (xor 1,2,4,8)
  float rinv[8];
#pragma unroll
  for (int j = 0; j < 8; ++j) {
    float s = lsum[j];
    s += __shfl_xor(s, 1, 32); s += __shfl_xor(s, 2, 32);
    s += __shfl_xor(s, 4, 32); s += __shfl_xor(s, 8, 32);
    rinv[j] = 1.0f / s;
  }

  // ====== Pass 2: recompute scores, write normalized weights, PV WMMA ======
  v8f xacc[8];
#pragma unroll
  for (int dt = 0; dt < 8; ++dt) xacc[dt] = (v8f){};
  uint8_t* prow = smem + PT_OFF + w * (16 * PT_PITCH);   // wave-private P tile

  for (int it = 0; it < Sk / BN; ++it) {
    const int s0 = it * BN;
    __syncthreads();
    if (t < 32) tdm_mask(Mb + s0, mask_lds);
    stage_k(smem, Kb + (size_t)s0 * Dh, t);
    stage_v(smem, Vb + (size_t)s0 * Dh, t);
    if (t < 32) __builtin_amdgcn_s_wait_tensorcnt(0);
    __syncthreads();

    float* wout = Wo + ((size_t)bn * Lq + qrow + 8 * h) * Sk + s0;
#pragma unroll
    for (int nt = 0; nt < 4; ++nt) {
      v8f c = {};
      const uint8_t* krow = smem + KT_OFF + (nt * 16 + ln) * KT_PITCH + 16 * h;
#pragma unroll
      for (int kc = 0; kc < 4; ++kc) {
        V16 ub;
        ub.q[0] = *(const uint4*)(krow + kc * 64);
        ub.q[1] = *(const uint4*)(krow + kc * 64 + 32);
        c = __builtin_amdgcn_wmma_f32_16x16x32_f16(false, qa[kc], false, ub.v,
                                                   (short)0, c, false, false);
      }
      const uint8_t* mp = smem + MASK_OFF + (w * 16 + 8 * h) * BN + nt * 16 + ln;
#pragma unroll
      for (int j = 0; j < 8; ++j) {
        const float wgt = mp[j * BN] ? 0.f : __expf(c[j] * SCALE) * rinv[j];
        wout[(size_t)j * Sk + nt * 16 + ln] = wgt;           // normalized weights
        *(h16*)(prow + (8 * h + j) * PT_PITCH + (nt * 16 + ln) * 2) = (h16)wgt;
      }
    }

    // x += P(16xBN) @ V(BNx128): A-frags from wave-private LDS (in-order DS),
    // B-frags from transposed V tile (two contiguous b128 reads).
#pragma unroll
    for (int c2 = 0; c2 < 2; ++c2) {
      V16 ua;
      const uint8_t* pr = prow + ln * PT_PITCH + c2 * 64 + 16 * h;
      ua.q[0] = *(const uint4*)(pr);
      ua.q[1] = *(const uint4*)(pr + 32);
#pragma unroll
      for (int dt = 0; dt < 8; ++dt) {
        V16 ub;
        const uint8_t* vr = smem + VT_OFF + (dt * 16 + ln) * VT_PITCH + c2 * 64 + 16 * h;
        ub.q[0] = *(const uint4*)(vr);
        ub.q[1] = *(const uint4*)(vr + 32);
        xacc[dt] = __builtin_amdgcn_wmma_f32_16x16x32_f16(false, ua.v, false, ub.v,
                                                          (short)0, xacc[dt], false, false);
      }
    }
  }

  // ---- write x (16 rows x 128 cols per wave) ----
  float* xo = Xo + ((size_t)bn * Lq + qrow + 8 * h) * Dh;
#pragma unroll
  for (int dt = 0; dt < 8; ++dt)
#pragma unroll
    for (int j = 0; j < 8; ++j)
      xo[(size_t)j * Dh + dt * 16 + ln] = xacc[dt][j];
}

extern "C" void kernel_launch(void* const* d_in, const int* in_sizes, int n_in,
                              void* d_out, int out_size, void* d_ws, size_t ws_size,
                              hipStream_t stream) {
  (void)in_sizes; (void)n_in; (void)out_size; (void)d_ws; (void)ws_size;
  const float*   Q  = (const float*)d_in[0];
  const float*   K  = (const float*)d_in[1];
  const float*   V  = (const float*)d_in[2];
  const uint8_t* M  = (const uint8_t*)d_in[3];   // jnp bool -> 1 byte/elem
  float* Xo = (float*)d_out;                               // x: N*L*D
  float* Wo = (float*)d_out + (size_t)Nb * Lq * Dh;        // attn_weights: N*L*S
  dim3 grid(Nb * (Lq / BM));                               // 256 blocks
  attn_fwd_kernel<<<grid, 256, 0, stream>>>(Q, K, V, M, Xo, Wo);
}
